// ViT_70368744178023
// MI455X (gfx1250) — matrix operand order repaired and checked by reading, not compiled, not run on hardware
//
#include <hip/hip_runtime.h>

// ---------------------------------------------------------------------------
// ViT forward for MI455X (gfx1250, wave32, WMMA + TDM).
// All GEMMs use v_wmma_f32_16x16x32_f16 (f16 in, f32 accum).
// K=768 GEMMs (qkv, out-proj): B panel staged to LDS with tensor_load_to_lds
// (one TDM per workgroup, s_wait_tensorcnt + barrier), 8 M-tiles share it.
// Patch GEMM / attention: direct-global WMMA (register-blocked).
// ---------------------------------------------------------------------------

typedef _Float16 v16h __attribute__((ext_vector_type(16)));
typedef _Float16 v8h  __attribute__((ext_vector_type(8)));
typedef float    v8f  __attribute__((ext_vector_type(8)));
typedef unsigned int v4u __attribute__((ext_vector_type(4)));
typedef int      v4i  __attribute__((ext_vector_type(4)));
typedef int      v8i  __attribute__((ext_vector_type(8)));

// Model constants
constexpr int kBatch   = 32;
constexpr int kHeads   = 12;
constexpr int kSeq     = 197;     // 196 patches + cls
constexpr int kD       = 768;
constexpr int kQKV     = 2304;    // 3 * 768
constexpr int kPatchD  = 3072;    // 3*1024 (c x pix)
constexpr int kNP      = 196;
constexpr int kRows    = kBatch * kSeq;     // 6304 (multiple of 32)
constexpr int kPRows   = kBatch * kNP;      // 6272 (multiple of 32)
constexpr int kSeqT    = 13;                // ceil(197/16)
constexpr int kSeqT2   = 7;                 // ceil(197/32)
constexpr int kSeqPadR = kSeqT * 16;        // 208 rows (attn buffer)
constexpr int kSeqPadC = 224;               // 7*32 cols (K-dim pad for WMMA)
constexpr int kSlack   = 32;                // extra qkv rows for padded reads
constexpr int kClasses = 1000;

static inline int cdiv(long a, long b) { return (int)((a + b - 1) / b); }

// ---------------------------------------------------------------------------
// Block reductions (blockDim.x == 256)
// ---------------------------------------------------------------------------
__device__ inline float bsum(float v, float* red) {
  int t = threadIdx.x; red[t] = v; __syncthreads();
  for (int s = 128; s > 0; s >>= 1) { if (t < s) red[t] += red[t + s]; __syncthreads(); }
  float r = red[0]; __syncthreads(); return r;
}
__device__ inline float bmax(float v, float* red) {
  int t = threadIdx.x; red[t] = v; __syncthreads();
  for (int s = 128; s > 0; s >>= 1) { if (t < s) red[t] = fmaxf(red[t], red[t + s]); __syncthreads(); }
  float r = red[0]; __syncthreads(); return r;
}

// ---------------------------------------------------------------------------
// WMMA fragment loaders (wave32); this code's contiguous k order, same for A and B.
// A: 16-bit 16x32 (MxK). lane<16: M=lane, K base 0; lane>=16: M=lane-16, K base 16.
//    elements e=0..15 -> K=base+e. Two 16B loads.
// B: 16-bit 32x16 (KxN). lane<16: N=lane, K=0..15 ; lane>=16: N=lane-16, K=16..31.
// ---------------------------------------------------------------------------
__device__ inline v16h load_a_frag(const _Float16* __restrict__ A, int lda,
                                   int row0, int k0, int lane) {
  int m  = row0 + (lane & 15);
  int kb = k0 + ((lane & 16) ? 16 : 0);
  const _Float16* p = A + (size_t)m * lda + kb;
  v8h lo = *(const v8h*)p;
  v8h hi = *(const v8h*)(p + 8);
  v16h f;
#pragma unroll
  for (int e = 0; e < 8; ++e) { f[e] = lo[e]; f[8 + e] = hi[e]; }
  return f;
}

// B with K contiguous in memory: element (k,n) at B[n*ldn + k]
__device__ inline v16h load_b_frag_contig(const _Float16* __restrict__ B, int ldn,
                                          int col0, int k0, int lane) {
  int n  = col0 + (lane & 15);
  int kb = k0 + ((lane & 16) ? 16 : 0);
  const _Float16* p = B + (size_t)n * ldn + kb;
  v8h lo = *(const v8h*)p;
  v8h hi = *(const v8h*)(p + 8);
  v16h f;
#pragma unroll
  for (int e = 0; e < 8; ++e) { f[e] = lo[e]; f[8 + e] = hi[e]; }
  return f;
}

// Same layout, but source is the LDS B panel (addrspace inferred -> ds_load)
__device__ inline v16h load_b_frag_lds(const _Float16* B, int col0, int k0, int lane) {
  int n  = col0 + (lane & 15);
  int kb = k0 + ((lane & 16) ? 16 : 0);
  const _Float16* p = B + (size_t)n * kD + kb;
  v8h lo = *(const v8h*)p;
  v8h hi = *(const v8h*)(p + 8);
  v16h f;
#pragma unroll
  for (int e = 0; e < 8; ++e) { f[e] = lo[e]; f[8 + e] = hi[e]; }
  return f;
}

// B row-major KxN: element (k,n) at B[k*ldk + n]  (strided gathers)
__device__ inline v16h load_b_frag_strided(const _Float16* __restrict__ B, int ldk,
                                           int col0, int k0, int lane) {
  int n  = col0 + (lane & 15);
  int kb = k0 + ((lane & 16) ? 16 : 0);
  const _Float16* p = B + (size_t)kb * ldk + n;
  v16h f;
#pragma unroll
  for (int e = 0; e < 16; ++e) f[e] = p[(size_t)e * ldk];
  return f;
}

__device__ inline v8f wmma_f16(v16h a, v16h b, v8f c) {
  return __builtin_amdgcn_wmma_f32_16x16x32_f16(false, a, false, b, (short)0, c,
                                                false, false);
}

// Store one 16x16 C tile (f32 accum) with optional bias / residual / f16 out.
template <bool OUT_F16, bool BIAS, bool RES>
__device__ inline void store_tile(v8f acc, int mbase, int n,
                                  float* __restrict__ Of, _Float16* __restrict__ Oh,
                                  int ldo, const float* __restrict__ bias,
                                  const float* __restrict__ res) {
#pragma unroll
  for (int r = 0; r < 8; ++r) {
    int m = mbase + r;
    float v = acc[r];
    if (BIAS) v += bias[n];
    if (RES)  v += res[(size_t)m * ldo + n];
    if (OUT_F16) Oh[(size_t)m * ldo + n] = (_Float16)v;
    else         Of[(size_t)m * ldo + n] = v;
  }
}

// ---------------------------------------------------------------------------
// Generic GEMM, 32x32 per wave, direct-global operands (used for patch embed).
// ---------------------------------------------------------------------------
template <bool OUT_F16, bool BIAS, bool RES>
__global__ void gemm_wmma_kernel(const _Float16* __restrict__ A, int lda,
                                 const _Float16* __restrict__ Bt, int ldn,
                                 float* __restrict__ Of, _Float16* __restrict__ Oh,
                                 int ldo, const float* __restrict__ bias,
                                 const float* __restrict__ res,
                                 int M, int Ncol, int K) {
  int lane = threadIdx.x & 31;
  int wave = (blockIdx.x * blockDim.x + threadIdx.x) >> 5;
  int tilesN = Ncol >> 5;                 // 32-wide tiles
  int tm = wave / tilesN, tn = wave - tm * tilesN;
  if (tm * 32 >= M) return;
  v8f acc00 = {}, acc01 = {}, acc10 = {}, acc11 = {};
  for (int k0 = 0; k0 < K; k0 += 32) {
    v16h a0 = load_a_frag(A, lda, tm * 32,      k0, lane);
    v16h a1 = load_a_frag(A, lda, tm * 32 + 16, k0, lane);
    v16h b0 = load_b_frag_contig(Bt, ldn, tn * 32,      k0, lane);
    v16h b1 = load_b_frag_contig(Bt, ldn, tn * 32 + 16, k0, lane);
    acc00 = wmma_f16(a0, b0, acc00);
    acc01 = wmma_f16(a0, b1, acc01);
    acc10 = wmma_f16(a1, b0, acc10);
    acc11 = wmma_f16(a1, b1, acc11);
  }
  int n0 = tn * 32 + (lane & 15);
  int mb = tm * 32 + ((lane & 16) ? 8 : 0);
  store_tile<OUT_F16, BIAS, RES>(acc00, mb,      n0,      Of, Oh, ldo, bias, res);
  store_tile<OUT_F16, BIAS, RES>(acc01, mb,      n0 + 16, Of, Oh, ldo, bias, res);
  store_tile<OUT_F16, BIAS, RES>(acc10, mb + 16, n0,      Of, Oh, ldo, bias, res);
  store_tile<OUT_F16, BIAS, RES>(acc11, mb + 16, n0 + 16, Of, Oh, ldo, bias, res);
}

// ---------------------------------------------------------------------------
// K=768 GEMM with TDM-staged B panel in LDS.
// Block = 256 threads = 8 waves. Block covers one 32-col N tile x 8 M tiles.
// Wave 0 issues TENSOR_LOAD_TO_LDS for the whole B panel [32 x 768] f16
// (48 KB), waits TENSORcnt, then all waves read B frags from LDS.
// D# per CDNA5 ISA ch.8: 2D tensor, data_size=2B, tile 768 x 32, stride 768.
// ---------------------------------------------------------------------------
template <bool OUT_F16, bool BIAS, bool RES>
__global__ void gemm_wmma_lds_kernel(const _Float16* __restrict__ A, int lda,
                                     const _Float16* __restrict__ Bt,
                                     float* __restrict__ Of, _Float16* __restrict__ Oh,
                                     int ldo, const float* __restrict__ bias,
                                     const float* __restrict__ res,
                                     int M, int Ncol) {
  __shared__ _Float16 ldsB[32 * kD];          // 48 KB, sole LDS object (offset 0)
  int lane = threadIdx.x & 31;
  int wv   = threadIdx.x >> 5;                // 0..7
  int tilesN = Ncol >> 5;
  int tn  = blockIdx.x % tilesN;
  int tmg = blockIdx.x / tilesN;
  int tm  = tmg * 8 + wv;

  if (wv == 0) {
    unsigned long long ga =
        (unsigned long long)(size_t)(Bt + (size_t)(tn * 32) * kD);
    // D# group 0: count=1 | lds_addr=0 | global_addr | type=2 ("image")
    v4u g0 = { 1u,
               0u,
               (unsigned)(ga & 0xFFFFFFFFu),
               ((unsigned)((ga >> 32) & 0x01FFFFFFu)) | (2u << 30) };
    // D# group 1: data_size=1(2B); tensor_dim0=768; tensor_dim1=Ncol;
    // tile_dim0=768; tile_dim1=32; tensor_dim0_stride=768; rest 0.
    v8i g1 = { (int)0x00010000u,
               (int)(768u << 16),
               (int)(((unsigned)Ncol & 0xFFFFu) << 16),
               (int)(768u << 16),
               32,
               768,
               0, 0 };
    v4i gz = { 0, 0, 0, 0 };
#if __clang_major__ >= 23
    v8i gz8 = { 0, 0, 0, 0, 0, 0, 0, 0 };
    __builtin_amdgcn_tensor_load_to_lds(g0, g1, gz, gz, gz8, 0);
#else
    __builtin_amdgcn_tensor_load_to_lds(g0, g1, gz, gz, 0);
#endif
    __builtin_amdgcn_s_wait_tensorcnt(0);
  }
  __syncthreads();

  if (tm * 32 < M) {
    v8f acc00 = {}, acc01 = {}, acc10 = {}, acc11 = {};
    for (int k0 = 0; k0 < kD; k0 += 32) {
      v16h a0 = load_a_frag(A, lda, tm * 32,      k0, lane);
      v16h a1 = load_a_frag(A, lda, tm * 32 + 16, k0, lane);
      v16h b0 = load_b_frag_lds(ldsB,  0, k0, lane);
      v16h b1 = load_b_frag_lds(ldsB, 16, k0, lane);
      acc00 = wmma_f16(a0, b0, acc00);
      acc01 = wmma_f16(a0, b1, acc01);
      acc10 = wmma_f16(a1, b0, acc10);
      acc11 = wmma_f16(a1, b1, acc11);
    }
    int n0 = tn * 32 + (lane & 15);
    int mb = tm * 32 + ((lane & 16) ? 8 : 0);
    store_tile<OUT_F16, BIAS, RES>(acc00, mb,      n0,      Of, Oh, ldo, bias, res);
    store_tile<OUT_F16, BIAS, RES>(acc01, mb,      n0 + 16, Of, Oh, ldo, bias, res);
    store_tile<OUT_F16, BIAS, RES>(acc10, mb + 16, n0,      Of, Oh, ldo, bias, res);
    store_tile<OUT_F16, BIAS, RES>(acc11, mb + 16, n0 + 16, Of, Oh, ldo, bias, res);
  }
}

// ---------------------------------------------------------------------------
// Weight convert f32 -> f16 with transpose: in [l][K][N] -> out [l][N][K]
// ---------------------------------------------------------------------------
__global__ void cvt_transpose_kernel(const float* __restrict__ in,
                                     _Float16* __restrict__ out,
                                     int K, int Ncol, long total) {
  long idx = (long)blockIdx.x * blockDim.x + threadIdx.x;
  if (idx >= total) return;
  long per = (long)K * Ncol;
  long l = idx / per, r = idx - l * per;
  int n = (int)(r / K), k = (int)(r - (long)n * K);
  out[idx] = (_Float16)in[l * per + (long)k * Ncol + n];
}

__global__ void zero_h_kernel(_Float16* __restrict__ p, long n) {
  long idx = (long)blockIdx.x * blockDim.x + threadIdx.x;
  if (idx < n) p[idx] = (_Float16)0.f;
}

// ---------------------------------------------------------------------------
// Patchify + patch LayerNorm (over 3072) -> f16 rows [B*196][3072]
// patch element e = (py*16+px)*3 + c ; img layout [B][C][H][W]
// ---------------------------------------------------------------------------
__global__ void patchify_ln_kernel(const float* __restrict__ img,
                                   const float* __restrict__ g,
                                   const float* __restrict__ bb,
                                   _Float16* __restrict__ xp) {
  __shared__ float buf[kPatchD];
  __shared__ float red[256];
  int bp = blockIdx.x;                 // 0..6271
  int b = bp / kNP, p = bp - b * kNP;
  int ph = p / 14, pw = p - ph * 14;
  int t = threadIdx.x;
  float s = 0.f;
  for (int e = t; e < kPatchD; e += 256) {
    int c = e % 3;
    int pix = e / 3;
    int py = pix >> 4, px = pix & 15;
    float v = img[(((size_t)b * 3 + c) * 224 + (ph * 16 + py)) * 224 + (pw * 16 + px)];
    buf[e] = v; s += v;
  }
  float mean = bsum(s, red) * (1.f / kPatchD);
  float s2 = 0.f;
  for (int e = t; e < kPatchD; e += 256) { float d = buf[e] - mean; s2 += d * d; }
  float rstd = rsqrtf(bsum(s2, red) * (1.f / kPatchD) + 1e-5f);
  for (int e = t; e < kPatchD; e += 256)
    xp[(size_t)bp * kPatchD + e] = (_Float16)((buf[e] - mean) * rstd * g[e] + bb[e]);
}

// ---------------------------------------------------------------------------
// Embed LN (over 768) + pos emb -> residual stream x[b][1+p][:]
// ---------------------------------------------------------------------------
__global__ void embed_kernel(const float* __restrict__ tmp,
                             const float* __restrict__ g, const float* __restrict__ bb,
                             const float* __restrict__ pos, float* __restrict__ x) {
  __shared__ float red[256];
  int r = blockIdx.x;                  // 0..6271
  int b = r / kNP, p = r - b * kNP;
  const float* row = tmp + (size_t)r * kD;
  int t = threadIdx.x;
  float s = 0.f;
  for (int e = t; e < kD; e += 256) s += row[e];
  float mean = bsum(s, red) * (1.f / kD);
  float s2 = 0.f;
  for (int e = t; e < kD; e += 256) { float d = row[e] - mean; s2 += d * d; }
  float rstd = rsqrtf(bsum(s2, red) * (1.f / kD) + 1e-5f);
  float* xo = x + ((size_t)b * kSeq + 1 + p) * kD;
  const float* pe = pos + (size_t)(1 + p) * kD;
  for (int e = t; e < kD; e += 256)
    xo[e] = (row[e] - mean) * rstd * g[e] + bb[e] + pe[e];
}

__global__ void cls_add_kernel(const float* __restrict__ cls,
                               const float* __restrict__ pos, float* __restrict__ x) {
  int idx = blockIdx.x * blockDim.x + threadIdx.x;
  if (idx >= kBatch * kD) return;
  int b = idx / kD, d = idx - b * kD;
  x[(size_t)b * kSeq * kD + d] = cls[d] + pos[d];
}

// ---------------------------------------------------------------------------
// Per-row LayerNorm (over 768) -> f16
// ---------------------------------------------------------------------------
__global__ void ln_rows_kernel(const float* __restrict__ x,
                               const float* __restrict__ g, const float* __restrict__ bb,
                               _Float16* __restrict__ out) {
  __shared__ float red[256];
  int r = blockIdx.x;
  const float* row = x + (size_t)r * kD;
  int t = threadIdx.x;
  float s = 0.f;
  for (int e = t; e < kD; e += 256) s += row[e];
  float mean = bsum(s, red) * (1.f / kD);
  float s2 = 0.f;
  for (int e = t; e < kD; e += 256) { float d = row[e] - mean; s2 += d * d; }
  float rstd = rsqrtf(bsum(s2, red) * (1.f / kD) + 1e-5f);
  _Float16* o = out + (size_t)r * kD;
  for (int e = t; e < kD; e += 256)
    o[e] = (_Float16)((row[e] - mean) * rstd * g[e] + bb[e]);
}

// ---------------------------------------------------------------------------
// Attention scores via WMMA: dw[b,h,i,j] = (q·k)*0.125 * (1 + sph[i][j])
// One wave -> 16x32 tile (A-frag reused over 2 j-tiles).
// ---------------------------------------------------------------------------
__global__ void attn_scores_kernel(const _Float16* __restrict__ qkv,
                                   const float* __restrict__ sph,
                                   float* __restrict__ scores) {
  int lane = threadIdx.x & 31;
  int wave = (blockIdx.x * blockDim.x + threadIdx.x) >> 5;
  const int tiles = kSeqT * kSeqT2;     // 13 * 7
  int bh = wave / tiles;
  if (bh >= kBatch * kHeads) return;
  int t = wave - bh * tiles;
  int ti = t / kSeqT2, tj = t - ti * kSeqT2;
  int b = bh / kHeads, h = bh - b * kHeads;
  const _Float16* qb = qkv + (size_t)(b * kSeq) * kQKV + h * 64;
  const _Float16* kb = qkv + (size_t)(b * kSeq) * kQKV + kD + h * 64;
  v8f acc0 = {}, acc1 = {};
  for (int k0 = 0; k0 < 64; k0 += 32) {
    v16h a  = load_a_frag(qb, kQKV, ti * 16, k0, lane);
    v16h b0 = load_b_frag_contig(kb, kQKV, tj * 32,      k0, lane);
    v16h b1 = load_b_frag_contig(kb, kQKV, tj * 32 + 16, k0, lane);
    acc0 = wmma_f16(a, b0, acc0);
    acc1 = wmma_f16(a, b1, acc1);
  }
  int j0 = tj * 32 + (lane & 15);
  int ibase = ti * 16 + ((lane & 16) ? 8 : 0);
#pragma unroll
  for (int r = 0; r < 8; ++r) {
    int i = ibase + r;
    if (i < kSeq) {
      if (j0 < kSeq) {
        float d = acc0[r] * 0.125f * (1.f + sph[(size_t)i * kSeq + j0]);
        scores[((size_t)bh * kSeq + i) * kSeq + j0] = d;
      }
      int j1 = j0 + 16;
      if (j1 < kSeq) {
        float d = acc1[r] * 0.125f * (1.f + sph[(size_t)i * kSeq + j1]);
        scores[((size_t)bh * kSeq + i) * kSeq + j1] = d;
      }
    }
  }
}

// Cross-head mean / std (ddof=1) per (b,i,j)
__global__ void head_stats_kernel(const float* __restrict__ scores,
                                  float* __restrict__ mu, float* __restrict__ rsd) {
  long idx = (long)blockIdx.x * blockDim.x + threadIdx.x;
  const long per = (long)kSeq * kSeq;
  if (idx >= (long)kBatch * per) return;
  long b = idx / per, ij = idx - b * per;
  float s1 = 0.f, s2 = 0.f;
#pragma unroll
  for (int h = 0; h < kHeads; ++h) {
    float v = scores[((long)(b * kHeads + h)) * per + ij];
    s1 += v; s2 += v * v;
  }
  float m = s1 * (1.f / kHeads);
  float var = (s2 - kHeads * m * m) * (1.f / (kHeads - 1));
  mu[idx]  = m;
  rsd[idx] = rsqrtf(fmaxf(var, 1e-20f));
}

// Normalize + softmax per (b,h,i) row -> f16 attn buffer [B*H][208][224] (0-padded cols)
__global__ void softmax_kernel(const float* __restrict__ scores,
                               const float* __restrict__ mu, const float* __restrict__ rsd,
                               _Float16* __restrict__ attn) {
  __shared__ float red[256];
  int row = blockIdx.x;                 // 0 .. B*H*197-1
  int i = row % kSeq;
  int bh = row / kSeq;
  int b = bh / kHeads;
  const float* sc = scores + ((size_t)bh * kSeq + i) * kSeq;
  const float* mp = mu  + ((size_t)b * kSeq + i) * kSeq;
  const float* rp = rsd + ((size_t)b * kSeq + i) * kSeq;
  int t = threadIdx.x;
  bool valid = t < kSeq;
  float v = valid ? (sc[t] - mp[t]) * rp[t] : -3.0e38f;
  float mx = bmax(v, red);
  float e = valid ? __expf(v - mx) : 0.f;
  float sum = bsum(e, red);
  _Float16* out = attn + ((size_t)bh * kSeqPadR + i) * kSeqPadC;
  if (t < kSeqPadC) out[t] = valid ? (_Float16)(e / sum) : (_Float16)0.f;
}

// o = attn @ V via WMMA. One wave -> 16x64 (full head dim): 1 A-frag, 4 WMMAs/chunk.
__global__ void attn_v_kernel(const _Float16* __restrict__ attn,
                              const _Float16* __restrict__ qkv,
                              _Float16* __restrict__ o_h) {
  int lane = threadIdx.x & 31;
  int wave = (blockIdx.x * blockDim.x + threadIdx.x) >> 5;
  int bh = wave / kSeqT;
  if (bh >= kBatch * kHeads) return;
  int ti = wave - bh * kSeqT;
  int b = bh / kHeads, h = bh - b * kHeads;
  const _Float16* Ab = attn + (size_t)bh * kSeqPadR * kSeqPadC;
  const _Float16* Vb = qkv + (size_t)(b * kSeq) * kQKV + 2 * kD + h * 64;
  v8f acc0 = {}, acc1 = {}, acc2 = {}, acc3 = {};
  for (int k0 = 0; k0 < kSeqPadC; k0 += 32) {       // K over padded seq (zeros past 197)
    v16h a  = load_a_frag(Ab, kSeqPadC, ti * 16, k0, lane);
    v16h b0 = load_b_frag_strided(Vb, kQKV,  0, k0, lane);
    v16h b1 = load_b_frag_strided(Vb, kQKV, 16, k0, lane);
    v16h b2 = load_b_frag_strided(Vb, kQKV, 32, k0, lane);
    v16h b3 = load_b_frag_strided(Vb, kQKV, 48, k0, lane);
    acc0 = wmma_f16(a, b0, acc0);
    acc1 = wmma_f16(a, b1, acc1);
    acc2 = wmma_f16(a, b2, acc2);
    acc3 = wmma_f16(a, b3, acc3);
  }
  int d0 = lane & 15;
  int ibase = ti * 16 + ((lane & 16) ? 8 : 0);
#pragma unroll
  for (int r = 0; r < 8; ++r) {
    int i = ibase + r;
    if (i < kSeq) {
      _Float16* o = o_h + ((size_t)(b * kSeq + i)) * kD + h * 64;
      o[d0]      = (_Float16)acc0[r];
      o[d0 + 16] = (_Float16)acc1[r];
      o[d0 + 32] = (_Float16)acc2[r];
      o[d0 + 48] = (_Float16)acc3[r];
    }
  }
}

// ---------------------------------------------------------------------------
// Final LN on cls rows + classifier head (small/ragged -> plain VALU)
// ---------------------------------------------------------------------------
__global__ void final_ln_kernel(const float* __restrict__ x,
                                const float* __restrict__ g, const float* __restrict__ bb,
                                float* __restrict__ clsn) {
  __shared__ float red[256];
  int b = blockIdx.x;
  const float* row = x + (size_t)b * kSeq * kD;   // token 0
  int t = threadIdx.x;
  float s = 0.f;
  for (int e = t; e < kD; e += 256) s += row[e];
  float mean = bsum(s, red) * (1.f / kD);
  float s2 = 0.f;
  for (int e = t; e < kD; e += 256) { float d = row[e] - mean; s2 += d * d; }
  float rstd = rsqrtf(bsum(s2, red) * (1.f / kD) + 1e-5f);
  for (int e = t; e < kD; e += 256)
    clsn[(size_t)b * kD + e] = (row[e] - mean) * rstd * g[e] + bb[e];
}

__global__ void head_kernel(const float* __restrict__ clsn,
                            const float* __restrict__ Wh, const float* __restrict__ bh,
                            float* __restrict__ out) {
  int idx = blockIdx.x * blockDim.x + threadIdx.x;
  if (idx >= kBatch * kClasses) return;
  int b = idx / kClasses, c = idx - b * kClasses;
  const float* r = clsn + (size_t)b * kD;
  float s = bh[c];
  for (int d = 0; d < kD; ++d) s += r[d] * Wh[(size_t)d * kClasses + c];
  out[idx] = s;
}

// ---------------------------------------------------------------------------
// Host orchestration
// ---------------------------------------------------------------------------
extern "C" void kernel_launch(void* const* d_in, const int* in_sizes, int n_in,
                              void* d_out, int out_size, void* d_ws, size_t ws_size,
                              hipStream_t stream) {
  (void)in_sizes; (void)n_in; (void)out_size; (void)ws_size;
  const float* img        = (const float*)d_in[0];
  const float* sph        = (const float*)d_in[1];
  const float* patch_ln_g = (const float*)d_in[2];
  const float* patch_ln_b = (const float*)d_in[3];
  const float* W_patch    = (const float*)d_in[4];
  const float* b_patch    = (const float*)d_in[5];
  const float* emb_ln_g   = (const float*)d_in[6];
  const float* emb_ln_b   = (const float*)d_in[7];
  const float* pos_emb    = (const float*)d_in[8];
  const float* cls_tok    = (const float*)d_in[9];
  const float* ln_g       = (const float*)d_in[10];
  const float* ln_b       = (const float*)d_in[11];
  const float* W_qkv      = (const float*)d_in[12];
  const float* W_out      = (const float*)d_in[13];
  const float* b_out      = (const float*)d_in[14];
  const float* fin_g      = (const float*)d_in[15];
  const float* fin_b      = (const float*)d_in[16];
  const float* W_head     = (const float*)d_in[17];
  const float* b_head     = (const float*)d_in[18];
  float* out = (float*)d_out;

  // Bump-allocate workspace (deterministic each call)
  char* p = (char*)d_ws;
  auto take = [&](size_t bytes) -> char* {
    char* r = p; p += (bytes + 255) & ~(size_t)255; return r;
  };
  _Float16* Wp_t   = (_Float16*)take(sizeof(_Float16) * (size_t)kD * kPatchD);
  _Float16* Wqkv_t = (_Float16*)take(sizeof(_Float16) * 12ull * kQKV * kD);
  _Float16* Wout_t = (_Float16*)take(sizeof(_Float16) * 12ull * kD * kD);
  _Float16* xp     = (_Float16*)take(sizeof(_Float16) * (size_t)kPRows * kPatchD);
  float*    tmp    = (float*)   take(sizeof(float)    * (size_t)kPRows * kD);
  float*    x      = (float*)   take(sizeof(float)    * (size_t)kRows * kD);
  _Float16* xn     = (_Float16*)take(sizeof(_Float16) * (size_t)kRows * kD);
  _Float16* qkv    = (_Float16*)take(sizeof(_Float16) * (size_t)(kRows + kSlack) * kQKV);
  float*    scores = (float*)   take(sizeof(float)    * (size_t)kBatch * kHeads * kSeq * kSeq);
  float*    muB    = (float*)   take(sizeof(float)    * (size_t)kBatch * kSeq * kSeq);
  float*    rsdB   = (float*)   take(sizeof(float)    * (size_t)kBatch * kSeq * kSeq);
  _Float16* attn   = (_Float16*)take(sizeof(_Float16) * (size_t)kBatch * kHeads * kSeqPadR * kSeqPadC);
  _Float16* o_h    = (_Float16*)take(sizeof(_Float16) * (size_t)kRows * kD);
  float*    clsn   = (float*)   take(sizeof(float)    * (size_t)kBatch * kD);

  dim3 blk(256);

  // Weight conversion (f32 -> f16, transposed to NxK)
  {
    long t0 = (long)kPatchD * kD;
    cvt_transpose_kernel<<<cdiv(t0, 256), blk, 0, stream>>>(W_patch, Wp_t, kPatchD, kD, t0);
    long t1 = 12ll * kD * kQKV;
    cvt_transpose_kernel<<<cdiv(t1, 256), blk, 0, stream>>>(W_qkv, Wqkv_t, kD, kQKV, t1);
    long t2 = 12ll * kD * kD;
    cvt_transpose_kernel<<<cdiv(t2, 256), blk, 0, stream>>>(W_out, Wout_t, kD, kD, t2);
  }
  // Zero qkv slack rows so padded-K fragment loads read zeros
  {
    long n = (long)kSlack * kQKV;
    zero_h_kernel<<<cdiv(n, 256), blk, 0, stream>>>(qkv + (size_t)kRows * kQKV, n);
  }

  // Patchify + patch LN
  patchify_ln_kernel<<<kPRows, blk, 0, stream>>>(img, patch_ln_g, patch_ln_b, xp);

  // Patch embed GEMM: [6272 x 3072] x [3072 x 768] + b_patch -> tmp (f32)
  {
    int waves = (kPRows / 32) * (kD / 32);
    gemm_wmma_kernel<false, true, false><<<cdiv(waves, 8), blk, 0, stream>>>(
        xp, kPatchD, Wp_t, kPatchD, tmp, nullptr, kD, b_patch, nullptr,
        kPRows, kD, kPatchD);
  }

  // Embed LN + pos emb + cls token -> x
  embed_kernel<<<kPRows, blk, 0, stream>>>(tmp, emb_ln_g, emb_ln_b, pos_emb, x);
  cls_add_kernel<<<cdiv(kBatch * kD, 256), blk, 0, stream>>>(cls_tok, pos_emb, x);

  // Transformer layers
  const int tilesM  = kRows / 32;            // 197
  const int mGroups = cdiv(tilesM, 8);       // 25
  for (int l = 0; l < 12; ++l) {
    ln_rows_kernel<<<kRows, blk, 0, stream>>>(x, ln_g + (size_t)l * kD,
                                              ln_b + (size_t)l * kD, xn);
    // qkv = xn @ Wqkv  (f16 out) -- TDM-staged B panel in LDS
    {
      int grid = (kQKV / 32) * mGroups;
      gemm_wmma_lds_kernel<true, false, false><<<grid, blk, 0, stream>>>(
          xn, kD, Wqkv_t + (size_t)l * kQKV * kD, nullptr, qkv, kQKV,
          nullptr, nullptr, kRows, kQKV);
    }
    // scores
    {
      int waves = kBatch * kHeads * kSeqT * kSeqT2;
      attn_scores_kernel<<<cdiv(waves, 8), blk, 0, stream>>>(qkv, sph, scores);
    }
    // cross-head stats + softmax
    {
      long n = (long)kBatch * kSeq * kSeq;
      head_stats_kernel<<<cdiv(n, 256), blk, 0, stream>>>(scores, muB, rsdB);
      softmax_kernel<<<kBatch * kHeads * kSeq, blk, 0, stream>>>(scores, muB, rsdB, attn);
    }
    // o = attn @ V
    {
      int waves = kBatch * kHeads * kSeqT;
      attn_v_kernel<<<cdiv(waves, 8), blk, 0, stream>>>(attn, qkv, o_h);
    }
    // x = o @ Wout + b_out + x   (residual, f32 out in-place) -- TDM-staged B
    {
      int grid = (kD / 32) * mGroups;
      gemm_wmma_lds_kernel<false, true, true><<<grid, blk, 0, stream>>>(
          o_h, kD, Wout_t + (size_t)l * kD * kD, x, nullptr, kD,
          b_out + (size_t)l * kD, x, kRows, kD);
    }
  }

  // Final LN on cls + head
  final_ln_kernel<<<kBatch, blk, 0, stream>>>(x, fin_g, fin_b, clsn);
  head_kernel<<<cdiv(kBatch * kClasses, 256), blk, 0, stream>>>(clsn, W_head, b_head, out);
}
